// TorchaxBaseLinearLayerWithLoRA_74938589381271
// MI455X (gfx1250) — compile-verified, matrix operand order checked by reading, not run on hardware
//
#include <hip/hip_runtime.h>

// Problem dims (fixed by the reference)
constexpr int T = 8192;
constexpr int D = 4096;
constexpr int O = 4096;
constexpr int L = 8;
constexpr int R = 16;
constexpr int KA = D + L * R;          // 4224 augmented K (base + flattened LoRA)
constexpr int KA8 = KA / 8;            // uint4 (8 halves) stride

// WMMA vector types (probe-confirmed signatures for gfx1250)
typedef __bf16 v16bf __attribute__((ext_vector_type(16)));
typedef float  v8f   __attribute__((ext_vector_type(8)));

// GEMM tiling
#define BM 128
#define BN 128
#define BK 32
#define LDSP 40   // padded row stride in halves (80 bytes) to spread LDS banks

// ---------------------------------------------------------------------------
// gfx1250 async global->LDS path (ASYNCcnt). Signature per hipcc diagnostic:
// param0 = v4i addrspace(1)* (global src), param1 = LDS dst, then imm, cpol.
// Guarded: falls back to global_load_b128 + ds_store_b128 staging.
// ---------------------------------------------------------------------------
#if __has_builtin(__builtin_amdgcn_global_load_async_to_lds_b128)
#define USE_ASYNC_LDS 1
typedef int v4i __attribute__((vector_size(16)));
__device__ __forceinline__ void copy16_g2l(const void* g, void* l) {
  __builtin_amdgcn_global_load_async_to_lds_b128(
      (__attribute__((address_space(1))) v4i*)g,
      (__attribute__((address_space(3))) v4i*)l,
      /*imm offset=*/0, /*cpol=*/0);
}
__device__ __forceinline__ void wait_lds_fill() {
  asm volatile("s_wait_asynccnt 0" ::: "memory");
}
#else
__device__ __forceinline__ void copy16_g2l(const void* g, void* l) {
  *(uint4*)l = *(const uint4*)g;
}
__device__ __forceinline__ void wait_lds_fill() {}
#endif

__device__ __forceinline__ unsigned short f2bf(float f) {
  unsigned int u = __float_as_uint(f);
  // round-to-nearest-even truncation to bf16
  unsigned int r = (u + 0x7FFFu + ((u >> 16) & 1u)) >> 16;
  return (unsigned short)r;
}

__device__ __forceinline__ v16bf make_frag(uint4 lo, uint4 hi) {
  union { uint4 u[2]; v16bf v; } un;
  un.u[0] = lo; un.u[1] = hi;
  return un.v;
}

// ---------------------------------------------------------------------------
// Kernel 1: pack Wb[O, KA] = [ bf16(weight) | bf16(lora_b flattened) ]
// ---------------------------------------------------------------------------
__global__ __launch_bounds__(256) void pack_w_kernel(
    const float* __restrict__ w, const float* __restrict__ lb,
    unsigned short* __restrict__ wb) {
  int i = blockIdx.x * 256 + threadIdx.x;        // i in [0, O*KA)
  int o = i / KA;
  int k = i - o * KA;
  float v;
  if (k < D) {
    v = w[(size_t)o * D + k];
  } else {
    int lr = k - D;
    int l = lr >> 4, r = lr & 15;
    v = lb[((size_t)l * O + o) * R + r];
  }
  wb[i] = f2bf(v);
}

// ---------------------------------------------------------------------------
// Kernel 2: one block per token.
//   Xb[t, 0:4096]       = bf16(x[t,:])
//   Xb[t, 4096 + l*R+r] = (l == idx[t]) ? bf16(dot(x[t,:], lora_a[idx[t]][r,:])) : 0
// lora_a is 2 MB -> resident in L2 across all 8192 blocks.
// ---------------------------------------------------------------------------
__global__ __launch_bounds__(256) void pack_x_shrink_kernel(
    const float* __restrict__ x, const float* __restrict__ la,
    const int* __restrict__ idx, unsigned short* __restrict__ xb) {
  int t   = blockIdx.x;
  int tid = threadIdx.x;
  const float* xr = x + (size_t)t * D;
  unsigned short* xo = xb + (size_t)t * KA;

  // bf16 conversion of the token row (coalesced)
  for (int k = tid; k < D; k += 256) xo[k] = f2bf(xr[k]);

  // shrink: r = tid>>4 (rank), g = tid&15 (d-group) -> contiguous 64B reads
  int r = tid >> 4;
  int g = tid & 15;
  int li = idx[t];
  const float* ar = la + ((size_t)li * R + r) * D;
  float p = 0.f;
  for (int d = g; d < D; d += 16) p += xr[d] * ar[d];

  __shared__ float red[256];
  red[tid] = p;
  __syncthreads();
  if (tid < R) {
    float s = 0.f;
    #pragma unroll
    for (int gg = 0; gg < 16; ++gg) s += red[tid * 16 + gg];
    unsigned short sb = f2bf(s);
    #pragma unroll
    for (int l = 0; l < L; ++l)
      xo[D + l * R + tid] = (l == li) ? sb : (unsigned short)0;
  }
}

// ---------------------------------------------------------------------------
// Kernel 3: bf16 WMMA GEMM  out[T,O] = Xb[T,KA] * Wb[O,KA]^T  (f32 accumulate)
// 256 threads = 8 wave32; block tile 128x128, K-step 32.
// Double-buffered LDS with *compile-time* buffer bases (K loop unrolled by 2),
// tile fill via async global->LDS b128 (ASYNCcnt) when available.
// Wave w: waveM = w>>2 (2 groups of 64 rows), waveN = w&3 (4 groups of 32 cols)
// -> 4x2 grid of 16x16 accumulators per wave, 8 WMMAs per k-step.
// ---------------------------------------------------------------------------
__global__ __launch_bounds__(256) void lora_gemm_kernel(
    const unsigned short* __restrict__ xb, const unsigned short* __restrict__ wb,
    float* __restrict__ out) {
  __shared__ __align__(16) unsigned short sA0[BM * LDSP];
  __shared__ __align__(16) unsigned short sA1[BM * LDSP];
  __shared__ __align__(16) unsigned short sB0[BN * LDSP];
  __shared__ __align__(16) unsigned short sB1[BN * LDSP];

  const int tid   = threadIdx.x;
  const int lane  = tid & 31;
  const int wave  = tid >> 5;
  const int waveM = wave >> 2;      // 0..1
  const int waveN = wave & 3;       // 0..3
  const int rowBase = blockIdx.y * BM;   // token rows
  const int colBase = blockIdx.x * BN;   // output cols

  const uint4* gA = (const uint4*)xb;
  const uint4* gB = (const uint4*)wb;

  v8f acc[4][2];
  #pragma unroll
  for (int mi = 0; mi < 4; ++mi)
    #pragma unroll
    for (int ni = 0; ni < 2; ++ni)
      acc[mi][ni] = (v8f){0.f, 0.f, 0.f, 0.f, 0.f, 0.f, 0.f, 0.f};

  // Tile fill: 512 16-byte chunks per operand; exactly 2 statically indexed
  // chunks per thread (no divergence, no loop).
  const int fillRow = tid >> 2;       // 0..63
  const int fillC4  = tid & 3;        // chunk within the 32-wide K slice
  const int fillKo  = fillC4 * 8;     // halves
  auto load_tiles = [&](unsigned short* dA, unsigned short* dB, int kBase) {
    const size_t k8 = (size_t)(kBase >> 3) + fillC4;
    #pragma unroll
    for (int h = 0; h < 2; ++h) {
      const int row = fillRow + h * 64;
      copy16_g2l(&gA[(size_t)(rowBase + row) * KA8 + k8],
                 &dA[row * LDSP + fillKo]);
      copy16_g2l(&gB[(size_t)(colBase + row) * KA8 + k8],
                 &dB[row * LDSP + fillKo]);
    }
  };

  // Fragment gather per the CDNA5 16-bit A/B striping:
  // lane<16: K 0-7 (vgpr0-3) and 16-23 (vgpr4-7); lane>=16: K 8-15 and 24-31
  const int fr = lane & 15;
  const int hs = lane >> 4;

  auto compute = [&](const unsigned short* dA, const unsigned short* dB) {
    v16bf aF[4], bF[2];
    #pragma unroll
    for (int mi = 0; mi < 4; ++mi) {
      const unsigned short* p = &dA[(waveM * 64 + mi * 16 + fr) * LDSP + hs * 8];
      aF[mi] = make_frag(*(const uint4*)p, *(const uint4*)(p + 16));
    }
    #pragma unroll
    for (int ni = 0; ni < 2; ++ni) {
      const unsigned short* p = &dB[(waveN * 32 + ni * 16 + fr) * LDSP + hs * 8];
      bF[ni] = make_frag(*(const uint4*)p, *(const uint4*)(p + 16));
    }
    #pragma unroll
    for (int mi = 0; mi < 4; ++mi)
      #pragma unroll
      for (int ni = 0; ni < 2; ++ni)
        acc[mi][ni] = __builtin_amdgcn_wmma_f32_16x16x32_bf16(
            false, aF[mi], false, bF[ni], (short)0, acc[mi][ni], false, false);
  };

  constexpr int KSTEPS = KA / BK;   // 132 (even)

  load_tiles(sA0, sB0, 0);
  wait_lds_fill();
  __syncthreads();

  #pragma unroll 1
  for (int ks = 0; ks < KSTEPS - 2; ks += 2) {
    load_tiles(sA1, sB1, (ks + 1) * BK);
    compute(sA0, sB0);
    wait_lds_fill();
    __syncthreads();
    load_tiles(sA0, sB0, (ks + 2) * BK);
    compute(sA1, sB1);
    wait_lds_fill();
    __syncthreads();
  }
  // tail: buf0 holds step KSTEPS-2
  load_tiles(sA1, sB1, (KSTEPS - 1) * BK);
  compute(sA0, sB0);
  wait_lds_fill();
  __syncthreads();
  compute(sA1, sB1);

  // Epilogue: C layout -> n = lane&15, m = vgpr + 8*(lane>=16).
  // Per-lane b32 stores are contiguous across lanes (coalesced lines).
  #pragma unroll
  for (int mi = 0; mi < 4; ++mi) {
    #pragma unroll
    for (int ni = 0; ni < 2; ++ni) {
      int col = colBase + waveN * 32 + ni * 16 + fr;
      int m0  = rowBase + waveM * 64 + mi * 16 + hs * 8;
      #pragma unroll
      for (int v = 0; v < 8; ++v)
        out[(size_t)(m0 + v) * O + col] = acc[mi][ni][v];
    }
  }
}

// ---------------------------------------------------------------------------
extern "C" void kernel_launch(void* const* d_in, const int* in_sizes, int n_in,
                              void* d_out, int out_size, void* d_ws, size_t ws_size,
                              hipStream_t stream) {
  const float* x   = (const float*)d_in[0];   // [T, D]
  const float* w   = (const float*)d_in[1];   // [O, D]
  const float* la  = (const float*)d_in[2];   // [L, R, D]
  const float* lb  = (const float*)d_in[3];   // [L, O, R]
  const int*   idx = (const int*)d_in[4];     // [T]
  float* out = (float*)d_out;                 // [T, O]

  unsigned short* xb   = (unsigned short*)d_ws;               // [T, KA] bf16
  unsigned short* wbuf = xb + (size_t)T * KA;                 // [O, KA] bf16

  pack_w_kernel<<<(O * KA) / 256, 256, 0, stream>>>(w, lb, wbuf);
  pack_x_shrink_kernel<<<T, 256, 0, stream>>>(x, la, idx, xb);

  dim3 grid(O / BN, T / BM);   // (32, 64); x-fastest -> column blocks share L2 row panel
  lora_gemm_kernel<<<grid, 256, 0, stream>>>(xb, wbuf, out);
}
